// RLModel_29308856828421
// MI455X (gfx1250) — matrix operand order repaired and checked by reading, not compiled, not run on hardware
//
#include <hip/hip_runtime.h>
#include <hip/hip_bf16.h>
#include <math.h>

// Problem constants (match reference)
#define B_    512
#define T_    512
#define OBS_  32
#define ACT_  16
#define H_    128
#define ROWS  (B_ * T_)            // 262144
#define GAMMA_ 0.99f
#define LBDA_  0.95f
#define EPS_BN_ 1e-5f
#define STATS_BLOCKS 256

typedef __attribute__((ext_vector_type(16))) __bf16          v16bf;
typedef __attribute__((ext_vector_type(16))) unsigned short  v16u;
typedef __attribute__((ext_vector_type(8)))  float           v8f;
typedef int av4i __attribute__((vector_size(16)));           // matches builtin's V4i

__device__ __forceinline__ unsigned short f2bf(float f) {
    unsigned int u = __builtin_bit_cast(unsigned int, f);
    unsigned int r = u + 0x7FFFu + ((u >> 16) & 1u);   // round-to-nearest-even
    return (unsigned short)(r >> 16);
}
__device__ __forceinline__ float bf2f(unsigned short h) {
    unsigned int u = ((unsigned int)h) << 16;
    return __builtin_bit_cast(float, u);
}

// ---------------------------------------------------------------------------
// Cooperative global->LDS copy of the fragment-packed weights.
// Uses the gfx1250 async-to-LDS path (global_load_async_to_lds_b128 +
// s_wait_asynccnt); falls back to a plain vectorized copy if unavailable.
// ---------------------------------------------------------------------------
__device__ __forceinline__ void copy_weights_to_lds(const unsigned short* __restrict__ Wf,
                                                    unsigned short* __restrict__ Wsh,
                                                    int nshorts)
{
    const int nvec = nshorts >> 3;                       // 16B chunks
#if defined(__AMDGCN__) && __has_builtin(__builtin_amdgcn_global_load_async_to_lds_b128)
    for (int i = threadIdx.x; i < nvec; i += blockDim.x) {
        __builtin_amdgcn_global_load_async_to_lds_b128(
            (__attribute__((address_space(1))) av4i*)(Wf + i * 8),
            (__attribute__((address_space(3))) av4i*)(Wsh + i * 8),
            0 /*offset*/, 0 /*cpol*/);
    }
#if __has_builtin(__builtin_amdgcn_s_wait_asynccnt)
    __builtin_amdgcn_s_wait_asynccnt(0);
#else
    asm volatile("s_wait_asynccnt 0x0" ::: "memory");
#endif
#else
    const uint4* src = (const uint4*)Wf;
    uint4*       dst = (uint4*)Wsh;
    for (int i = threadIdx.x; i < nvec; i += blockDim.x) dst[i] = src[i];
#endif
    __syncthreads();
}

// ---------------------------------------------------------------------------
// WMMA GEMM:  out[M x N] = A[M x K](bf16) * W(bf16, fragment-packed) + bias
// One wave computes TWO 16-row stripes (32 rows) across all N (NT tiles):
//  - B fragments staged in LDS (ds_load_b128), reused by both row stripes
//  - two independent accumulator chains hide WMMA RAW hazard NOPs
// A lane layout (16-bit A 16x32, ISA 7.12.2):
//   lane l holds row M = l&15; element e -> K = ((e>>3)<<4) + ((l>>4)*8) + (e&7)
// C/D layout: VGPR r, lane l -> M = r + (l>>4)*8, N = l&15.
// MODE: 0 = store bf16 (hidden, +bias), 1 = store f32 all cols (+bias),
//       2 = store f32 column 0 only (+bias)  (value head)
// ---------------------------------------------------------------------------
template<int K, int NT, int MODE>
__global__ __launch_bounds__(256)
void gemm_bf16_wmma(const unsigned short* __restrict__ A,
                    const unsigned short* __restrict__ Wf,
                    const float* __restrict__ bias,
                    void* __restrict__ outp)
{
    static_assert(K % 32 == 0, "K must be multiple of 32");
    constexpr int N = NT * 16;
    __shared__ __align__(16) unsigned short Wsh[K * N];   // <= 32 KB
    copy_weights_to_lds(Wf, Wsh, K * N);

    const int lane = threadIdx.x & 31;
    const int wave = threadIdx.x >> 5;
    const int mt0  = (blockIdx.x * 8 + wave) * 2;         // first 16-row tile
    if (mt0 * 16 >= ROWS) return;                         // wave-uniform guard

    v8f acc0[NT] = {};
    v8f acc1[NT] = {};
    const long aBase0 = (long)(mt0 * 16 + (lane & 15)) * K;
    const long aBase1 = aBase0 + (long)16 * K;
    const int  koffA  = (lane >> 4) * 8;

    #pragma unroll
    for (int kt = 0; kt < K / 32; ++kt) {
        v16u au0, au1;
        #pragma unroll
        for (int e = 0; e < 16; ++e) {
            const long o = kt * 32 + ((e >> 3) << 4) + koffA + (e & 7);
            au0[e] = A[aBase0 + o];
            au1[e] = A[aBase1 + o];
        }
        v16bf a0 = __builtin_bit_cast(v16bf, au0);
        v16bf a1 = __builtin_bit_cast(v16bf, au1);

        #pragma unroll
        for (int nt = 0; nt < NT; ++nt) {
            const unsigned short* wp = Wsh + (((kt * NT + nt) * 32 + lane) << 4);
            v16u bu;
            #pragma unroll
            for (int e = 0; e < 16; ++e) bu[e] = wp[e];
            v16bf b = __builtin_bit_cast(v16bf, bu);
            acc0[nt] = __builtin_amdgcn_wmma_f32_16x16x32_bf16(
                false, a0, false, b, (short)0, acc0[nt], false, false);
            acc1[nt] = __builtin_amdgcn_wmma_f32_16x16x32_bf16(
                false, a1, false, b, (short)0, acc1[nt], false, false);
        }
    }

    const int nlane = lane & 15;
    const int mhalf = (lane >> 4) << 3;
    #pragma unroll
    for (int nt = 0; nt < NT; ++nt) {
        const int n  = nt * 16 + nlane;
        const float bb = bias[n];
        #pragma unroll
        for (int r = 0; r < 8; ++r) {
            const long m0 = (long)mt0 * 16 + mhalf + r;
            const long m1 = m0 + 16;
            float v0 = acc0[nt][r] + bb;
            float v1 = acc1[nt][r] + bb;
            if constexpr (MODE == 0) {
                ((unsigned short*)outp)[m0 * N + n] = f2bf(v0);
                ((unsigned short*)outp)[m1 * N + n] = f2bf(v1);
            } else if constexpr (MODE == 1) {
                ((float*)outp)[m0 * N + n] = v0;
                ((float*)outp)[m1 * N + n] = v1;
            } else {
                if (n == 0) { ((float*)outp)[m0] = v0; ((float*)outp)[m1] = v1; }
            }
        }
    }
}

// ---- weight packer into WMMA fragment order (with K/N zero padding) -------
__global__ void pack_w_frag(const float* __restrict__ W, unsigned short* __restrict__ Wp,
                            int Kin, int Nin, int Kpad, int NT)
{
    const int total = Kpad * NT * 16;
    for (int idx = blockIdx.x * blockDim.x + threadIdx.x; idx < total;
         idx += gridDim.x * blockDim.x) {
        int e = idx & 15, lane = (idx >> 4) & 31;
        int q = idx >> 9;
        int nt = q % NT, kt = q / NT;
        int k = kt * 32 + ((e >> 3) << 4) + ((lane >> 4) << 3) + (e & 7);
        int n = nt * 16 + (lane & 15);
        float v = (k < Kin && n < Nin) ? W[(long)k * Nin + n] : 0.f;
        Wp[idx] = f2bf(v);
    }
}

__global__ void pad_bias16(const float* __restrict__ b, float* __restrict__ bp, int n)
{
    int i = threadIdx.x;
    if (i < 16) bp[i] = (i < n) ? b[i] : 0.f;
}

// ---- BN stats over input channels (obs ++ dones), one block per channel ---
__global__ __launch_bounds__(256)
void xs_stats_kernel(const float* __restrict__ obs, const unsigned char* __restrict__ dones,
                     float* __restrict__ mean, float* __restrict__ var)
{
    const int c = blockIdx.x;                        // 0..32
    float s = 0.f, s2 = 0.f;
    for (int r = threadIdx.x; r < ROWS; r += 256) {
        float v = (c < OBS_) ? obs[(long)r * OBS_ + c] : (dones[r] ? 1.f : 0.f);
        s += v; s2 += v * v;
    }
    __shared__ float sh[256], sh2[256];
    sh[threadIdx.x] = s; sh2[threadIdx.x] = s2; __syncthreads();
    for (int o = 128; o > 0; o >>= 1) {
        if (threadIdx.x < o) { sh[threadIdx.x] += sh[threadIdx.x + o];
                               sh2[threadIdx.x] += sh2[threadIdx.x + o]; }
        __syncthreads();
    }
    if (threadIdx.x == 0) {
        float m = sh[0] / (float)ROWS;
        mean[c] = m;
        var[c]  = sh2[0] / (float)ROWS - m * m;      // biased var (training BN)
    }
}

// ---- coalesced BN stats over bf16 hidden (C=128): partial then final ------
__global__ __launch_bounds__(256)
void bn_stats_partial(const unsigned short* __restrict__ X,
                      float* __restrict__ pm, float* __restrict__ pv)
{
    const int c  = threadIdx.x & (H_ - 1);
    const int rh = threadIdx.x >> 7;                 // 0 or 1
    const int rowsPerBlk = ROWS / STATS_BLOCKS;      // 1024
    const long rbase = (long)blockIdx.x * rowsPerBlk;
    float s = 0.f, s2 = 0.f;
    for (int i = 0; i < rowsPerBlk; i += 2) {
        float v = bf2f(X[(rbase + i + rh) * H_ + c]);
        s += v; s2 += v * v;
    }
    __shared__ float sh[256], sh2[256];
    sh[threadIdx.x] = s; sh2[threadIdx.x] = s2; __syncthreads();
    if (threadIdx.x < H_) {
        pm[(long)blockIdx.x * H_ + threadIdx.x] = sh[threadIdx.x] + sh[threadIdx.x + H_];
        pv[(long)blockIdx.x * H_ + threadIdx.x] = sh2[threadIdx.x] + sh2[threadIdx.x + H_];
    }
}

__global__ __launch_bounds__(256)
void bn_stats_final(const float* __restrict__ pm, const float* __restrict__ pv,
                    float* __restrict__ mean, float* __restrict__ var)
{
    const int c = blockIdx.x;                        // 0..127
    float s = 0.f, s2 = 0.f;
    for (int i = threadIdx.x; i < STATS_BLOCKS; i += 256) {
        s  += pm[(long)i * H_ + c];
        s2 += pv[(long)i * H_ + c];
    }
    __shared__ float sh[256], sh2[256];
    sh[threadIdx.x] = s; sh2[threadIdx.x] = s2; __syncthreads();
    for (int o = 128; o > 0; o >>= 1) {
        if (threadIdx.x < o) { sh[threadIdx.x] += sh[threadIdx.x + o];
                               sh2[threadIdx.x] += sh2[threadIdx.x + o]; }
        __syncthreads();
    }
    if (threadIdx.x == 0) {
        float m = sh[0] / (float)ROWS;
        mean[c] = m;
        var[c]  = sh2[0] / (float)ROWS - m * m;
    }
}

// ---- build normalized bf16 input, K padded 33 -> 64 -----------------------
__global__ void xs_pack_kernel(const float* __restrict__ obs, const unsigned char* __restrict__ dones,
                               const float* __restrict__ g, const float* __restrict__ bb,
                               const float* __restrict__ mean, const float* __restrict__ var,
                               unsigned short* __restrict__ A0)
{
    const long total = (long)ROWS * 64;
    for (long idx = (long)blockIdx.x * blockDim.x + threadIdx.x; idx < total;
         idx += (long)gridDim.x * blockDim.x) {
        int k = (int)(idx & 63);
        long row = idx >> 6;
        float y = 0.f;
        if (k < 33) {
            float x = (k < OBS_) ? obs[row * OBS_ + k] : (dones[row] ? 1.f : 0.f);
            y = (x - mean[k]) * rsqrtf(var[k] + EPS_BN_) * g[k] + bb[k];
        }
        A0[idx] = f2bf(y);
    }
}

// ---- in-place BN + ReLU on bf16 hidden ------------------------------------
__global__ void bn_relu_kernel(unsigned short* __restrict__ X,
                               const float* __restrict__ g, const float* __restrict__ bb,
                               const float* __restrict__ mean, const float* __restrict__ var)
{
    const long total = (long)ROWS * H_;
    for (long idx = (long)blockIdx.x * blockDim.x + threadIdx.x; idx < total;
         idx += (long)gridDim.x * blockDim.x) {
        int c = (int)(idx & (H_ - 1));
        float v = bf2f(X[idx]);
        float y = (v - mean[c]) * rsqrtf(var[c] + EPS_BN_) * g[c] + bb[c];
        X[idx] = f2bf(fmaxf(y, 0.f));
    }
}

// ---- reward variance (unbiased) -> scale ----------------------------------
__global__ __launch_bounds__(256)
void rvar_partial(const float* __restrict__ rewards, float* __restrict__ p1, float* __restrict__ p2)
{
    float s = 0.f, s2 = 0.f;
    for (int i = blockIdx.x * 256 + threadIdx.x; i < ROWS; i += 512 * 256) {
        float v = rewards[i]; s += v; s2 += v * v;
    }
    __shared__ float sh[256], sh2[256];
    sh[threadIdx.x] = s; sh2[threadIdx.x] = s2; __syncthreads();
    for (int o = 128; o > 0; o >>= 1) {
        if (threadIdx.x < o) { sh[threadIdx.x] += sh[threadIdx.x + o];
                               sh2[threadIdx.x] += sh2[threadIdx.x + o]; }
        __syncthreads();
    }
    if (threadIdx.x == 0) { p1[blockIdx.x] = sh[0]; p2[blockIdx.x] = sh2[0]; }
}

__global__ __launch_bounds__(256)
void rvar_final(const float* __restrict__ p1, const float* __restrict__ p2, float* __restrict__ scale)
{
    float s = 0.f, s2 = 0.f;
    for (int i = threadIdx.x; i < 512; i += 256) { s += p1[i]; s2 += p2[i]; }
    __shared__ float sh[256], sh2[256];
    sh[threadIdx.x] = s; sh2[threadIdx.x] = s2; __syncthreads();
    for (int o = 128; o > 0; o >>= 1) {
        if (threadIdx.x < o) { sh[threadIdx.x] += sh[threadIdx.x + o];
                               sh2[threadIdx.x] += sh2[threadIdx.x + o]; }
        __syncthreads();
    }
    if (threadIdx.x == 0) {
        float n = (float)ROWS;
        float var = (sh2[0] - sh[0] * sh[0] / n) / (n - 1.f);   // unbiased
        scale[0] = 1.f / fmaxf(sqrtf(var), 1e-8f);
    }
}

// ---- reverse-time scans ----------------------------------------------------
__global__ void scan_R_kernel(const float* __restrict__ rewards, const unsigned char* __restrict__ dones,
                              const float* __restrict__ value_old, const float* __restrict__ scale,
                              float* __restrict__ Rs)
{
    int b = blockIdx.x * blockDim.x + threadIdx.x;
    if (b >= B_) return;
    float sc = scale[0];
    float R = value_old[(long)b * T_ + (T_ - 1)];
    for (int t = T_ - 1; t >= 0; --t) {
        long i = (long)b * T_ + t;
        float rn = rewards[i] * sc;
        R = rn + GAMMA_ * (dones[i] ? 0.f : R);
        Rs[i] = R;
    }
}

__global__ void scan_gae_kernel(const float* __restrict__ rewards, const unsigned char* __restrict__ dones,
                                const float* __restrict__ value_old, const float* __restrict__ scale,
                                float* __restrict__ Agae)
{
    int b = blockIdx.x * blockDim.x + threadIdx.x;
    if (b >= B_) return;
    float sc = scale[0];
    float gae = 0.f;
    for (int t = T_ - 2; t >= 0; --t) {
        long i = (long)b * T_ + t;
        float nv = dones[i] ? 0.f : value_old[i + 1];
        float td = rewards[i] * sc + GAMMA_ * nv - value_old[i];
        gae = td + GAMMA_ * LBDA_ * (dones[i] ? 0.f : gae);
        Agae[(long)b * (T_ - 1) + t] = gae;
    }
}

// ---- log-softmax gather ----------------------------------------------------
__global__ void logp_kernel(const float* __restrict__ logits, const float* __restrict__ logits_old,
                            const int* __restrict__ actions,
                            float* __restrict__ lp, float* __restrict__ lpo)
{
    for (long row = (long)blockIdx.x * blockDim.x + threadIdx.x; row < ROWS;
         row += (long)gridDim.x * blockDim.x) {
        int a = actions[row];
        {
            const float* L = logits + row * ACT_;
            float mx = -3.4e38f;
            for (int j = 0; j < ACT_; ++j) mx = fmaxf(mx, L[j]);
            float s = 0.f;
            for (int j = 0; j < ACT_; ++j) s += expf(L[j] - mx);
            lp[row] = L[a] - (mx + logf(s));
        }
        {
            const float* L = logits_old + row * ACT_;
            float mx = -3.4e38f;
            for (int j = 0; j < ACT_; ++j) mx = fmaxf(mx, L[j]);
            float s = 0.f;
            for (int j = 0; j < ACT_; ++j) s += expf(L[j] - mx);
            lpo[row] = L[a] - (mx + logf(s));
        }
    }
}

// ---- loss reductions -------------------------------------------------------
__global__ __launch_bounds__(256)
void loss_partial_kernel(const float* __restrict__ lp, const float* __restrict__ lpo,
                         const float* __restrict__ Rs, const float* __restrict__ vpred,
                         const float* __restrict__ Agae,
                         float* __restrict__ pent, float* __restrict__ pval, float* __restrict__ pppo)
{
    float se = 0.f, sv = 0.f, sp = 0.f;
    for (long i = (long)blockIdx.x * 256 + threadIdx.x; i < ROWS; i += (long)512 * 256) {
        se += lp[i];
        float d = Rs[i] - vpred[i];
        sv += d * d;
        int t = (int)(i % T_);
        if (t < T_ - 1) {
            long b = i / T_;
            float A = Agae[b * (T_ - 1) + t];
            float r = expf(lp[i] - lpo[i]);
            // clip(r, 0.8, 0.2) collapses to the constant 0.2 (faithful to ref)
            sp += fminf(r * A, 0.2f * A);
        }
    }
    __shared__ float sh[256], sh2[256], sh3[256];
    sh[threadIdx.x] = se; sh2[threadIdx.x] = sv; sh3[threadIdx.x] = sp; __syncthreads();
    for (int o = 128; o > 0; o >>= 1) {
        if (threadIdx.x < o) { sh[threadIdx.x] += sh[threadIdx.x + o];
                               sh2[threadIdx.x] += sh2[threadIdx.x + o];
                               sh3[threadIdx.x] += sh3[threadIdx.x + o]; }
        __syncthreads();
    }
    if (threadIdx.x == 0) { pent[blockIdx.x] = sh[0]; pval[blockIdx.x] = sh2[0]; pppo[blockIdx.x] = sh3[0]; }
}

__global__ __launch_bounds__(256)
void loss_final_kernel(const float* __restrict__ pent, const float* __restrict__ pval,
                       const float* __restrict__ pppo, float* __restrict__ out)
{
    float se = 0.f, sv = 0.f, sp = 0.f;
    for (int i = threadIdx.x; i < 512; i += 256) { se += pent[i]; sv += pval[i]; sp += pppo[i]; }
    __shared__ float sh[256], sh2[256], sh3[256];
    sh[threadIdx.x] = se; sh2[threadIdx.x] = sv; sh3[threadIdx.x] = sp; __syncthreads();
    for (int o = 128; o > 0; o >>= 1) {
        if (threadIdx.x < o) { sh[threadIdx.x] += sh[threadIdx.x + o];
                               sh2[threadIdx.x] += sh2[threadIdx.x + o];
                               sh3[threadIdx.x] += sh3[threadIdx.x + o]; }
        __syncthreads();
    }
    if (threadIdx.x == 0) {
        out[0] = -(sh3[0] / (float)(B_ * (T_ - 1)));   // ppo_loss
        out[1] = sh2[0] / (float)ROWS;                 // value_loss
        out[2] = -(sh[0] / (float)ROWS);               // entropy_loss
    }
}

// ===========================================================================
extern "C" void kernel_launch(void* const* d_in, const int* in_sizes, int n_in,
                              void* d_out, int out_size, void* d_ws, size_t ws_size,
                              hipStream_t stream)
{
    (void)in_sizes; (void)n_in; (void)out_size; (void)ws_size;

    const float*         obs        = (const float*)d_in[0];
    const float*         rewards    = (const float*)d_in[1];
    const float*         value_old  = (const float*)d_in[2];
    const float*         logits_old = (const float*)d_in[3];
    const unsigned char* dones      = (const unsigned char*)d_in[4];
    const int*           actions    = (const int*)d_in[5];
    // param order: bn_in_g, bn_in_b, w1, b1, bn1_g, bn1_b, w2, b2, bn2_g, bn2_b, w3, b3
    const float* PP[12]; for (int i = 0; i < 12; ++i) PP[i] = (const float*)d_in[6  + i];
    const float* VP[12]; for (int i = 0; i < 12; ++i) VP[i] = (const float*)d_in[18 + i];
    float* out = (float*)d_out;

    // ---- workspace carve-up (~190 MB) ----
    char* ws = (char*)d_ws;
    size_t off = 0;
    auto carve = [&](size_t bytes) -> char* {
        char* p = ws + off; off += (bytes + 255) & ~(size_t)255; return p;
    };
    unsigned short* A0   = (unsigned short*)carve((size_t)ROWS * 64 * 2);
    unsigned short* Hb1  = (unsigned short*)carve((size_t)ROWS * H_ * 2);
    unsigned short* Hb2  = (unsigned short*)carve((size_t)ROWS * H_ * 2);
    float* Plog = (float*)carve((size_t)ROWS * ACT_ * 4);
    float* Vout = (float*)carve((size_t)ROWS * 4);
    float* Rs   = (float*)carve((size_t)ROWS * 4);
    float* Agae = (float*)carve((size_t)B_ * (T_ - 1) * 4);
    float* lp   = (float*)carve((size_t)ROWS * 4);
    float* lpo  = (float*)carve((size_t)ROWS * 4);
    unsigned short* Wp1 = (unsigned short*)carve(64 * 128 * 2);
    unsigned short* Wp2 = (unsigned short*)carve(128 * 128 * 2);
    unsigned short* Wp3 = (unsigned short*)carve(128 * 16 * 2);
    float* mean_in = (float*)carve(64 * 4);
    float* var_in  = (float*)carve(64 * 4);
    float* meanH   = (float*)carve(128 * 4);
    float* varH    = (float*)carve(128 * 4);
    float* scale   = (float*)carve(4);
    float* bpad    = (float*)carve(16 * 4);
    float* pmS = (float*)carve((size_t)STATS_BLOCKS * H_ * 4);
    float* pvS = (float*)carve((size_t)STATS_BLOCKS * H_ * 4);
    float* r1 = (float*)carve(512 * 4);
    float* r2 = (float*)carve(512 * 4);
    float* pe = (float*)carve(512 * 4);
    float* pv = (float*)carve(512 * 4);
    float* pq = (float*)carve(512 * 4);

    // ---- reward variance scale + input BN stats (shared by both MLPs) ----
    rvar_partial<<<512, 256, 0, stream>>>(rewards, r1, r2);
    rvar_final<<<1, 256, 0, stream>>>(r1, r2, scale);
    xs_stats_kernel<<<33, 256, 0, stream>>>(obs, dones, mean_in, var_in);

    const int GEMM_BLOCKS = ROWS / (32 * 8);   // 1024 (32 rows/wave, 8 waves/block)

    // =================== POLICY MLP ===================
    xs_pack_kernel<<<4096, 256, 0, stream>>>(obs, dones, PP[0], PP[1], mean_in, var_in, A0);
    pack_w_frag<<<32, 256, 0, stream>>>(PP[2], Wp1, 33, 128, 64, 8);
    gemm_bf16_wmma<64, 8, 0><<<GEMM_BLOCKS, 256, 0, stream>>>(A0, Wp1, PP[3], Hb1);
    bn_stats_partial<<<STATS_BLOCKS, 256, 0, stream>>>(Hb1, pmS, pvS);
    bn_stats_final<<<128, 256, 0, stream>>>(pmS, pvS, meanH, varH);
    bn_relu_kernel<<<4096, 256, 0, stream>>>(Hb1, PP[4], PP[5], meanH, varH);
    pack_w_frag<<<64, 256, 0, stream>>>(PP[6], Wp2, 128, 128, 128, 8);
    gemm_bf16_wmma<128, 8, 0><<<GEMM_BLOCKS, 256, 0, stream>>>(Hb1, Wp2, PP[7], Hb2);
    bn_stats_partial<<<STATS_BLOCKS, 256, 0, stream>>>(Hb2, pmS, pvS);
    bn_stats_final<<<128, 256, 0, stream>>>(pmS, pvS, meanH, varH);
    bn_relu_kernel<<<4096, 256, 0, stream>>>(Hb2, PP[8], PP[9], meanH, varH);
    pack_w_frag<<<8, 256, 0, stream>>>(PP[10], Wp3, 128, 16, 128, 1);
    gemm_bf16_wmma<128, 1, 1><<<GEMM_BLOCKS, 256, 0, stream>>>(Hb2, Wp3, PP[11], Plog);

    // =================== VALUE MLP ===================
    xs_pack_kernel<<<4096, 256, 0, stream>>>(obs, dones, VP[0], VP[1], mean_in, var_in, A0);
    pack_w_frag<<<32, 256, 0, stream>>>(VP[2], Wp1, 33, 128, 64, 8);
    gemm_bf16_wmma<64, 8, 0><<<GEMM_BLOCKS, 256, 0, stream>>>(A0, Wp1, VP[3], Hb1);
    bn_stats_partial<<<STATS_BLOCKS, 256, 0, stream>>>(Hb1, pmS, pvS);
    bn_stats_final<<<128, 256, 0, stream>>>(pmS, pvS, meanH, varH);
    bn_relu_kernel<<<4096, 256, 0, stream>>>(Hb1, VP[4], VP[5], meanH, varH);
    pack_w_frag<<<64, 256, 0, stream>>>(VP[6], Wp2, 128, 128, 128, 8);
    gemm_bf16_wmma<128, 8, 0><<<GEMM_BLOCKS, 256, 0, stream>>>(Hb1, Wp2, VP[7], Hb2);
    bn_stats_partial<<<STATS_BLOCKS, 256, 0, stream>>>(Hb2, pmS, pvS);
    bn_stats_final<<<128, 256, 0, stream>>>(pmS, pvS, meanH, varH);
    bn_relu_kernel<<<4096, 256, 0, stream>>>(Hb2, VP[8], VP[9], meanH, varH);
    pack_w_frag<<<8, 256, 0, stream>>>(VP[10], Wp3, 128, 1, 128, 1);
    pad_bias16<<<1, 16, 0, stream>>>(VP[11], bpad, 1);
    gemm_bf16_wmma<128, 1, 2><<<GEMM_BLOCKS, 256, 0, stream>>>(Hb2, Wp3, bpad, Vout);

    // =================== scans + losses ===================
    scan_R_kernel<<<2, 256, 0, stream>>>(rewards, dones, value_old, scale, Rs);
    scan_gae_kernel<<<2, 256, 0, stream>>>(rewards, dones, value_old, scale, Agae);
    logp_kernel<<<1024, 256, 0, stream>>>(Plog, logits_old, actions, lp, lpo);
    loss_partial_kernel<<<512, 256, 0, stream>>>(lp, lpo, Rs, Vout, Agae, pe, pv, pq);
    loss_final_kernel<<<1, 256, 0, stream>>>(pe, pv, pq, out);
}
